// OnLSTMCell_88124138979438
// MI455X (gfx1250) — matrix operand order repaired and checked by reading, not compiled, not run on hardware
//
#include <hip/hip_runtime.h>
#include <hip/hip_bf16.h>

// ---- problem dims (fixed by setup_inputs) ----
#define BSZ   8192
#define INSZ  1024
#define HID   1024
#define DIN   2048          // INSZ + HID
#define LVL   64
#define WROWS 4224          // 4*HID + 2*LVL

// main-GEMM tiling
#define BM    128           // rows per block
#define BN    32            // hidden cols per block (x4 gates)
#define KT    32            // K per step
#define NKT   (DIN / KT)    // 64 steps
#define LDSPAD 40           // 32 + 8 bf16 pad (80B row stride, conflict-free)

typedef __attribute__((ext_vector_type(16))) __bf16 v16bf;
typedef __attribute__((ext_vector_type(8)))  float  v8f;
typedef __attribute__((ext_vector_type(4)))  int    v4i;

// ---- CDNA5 async global->LDS path (ASYNCcnt), with portable fallback ----
#if defined(__has_builtin)
#if __has_builtin(__builtin_amdgcn_global_load_async_to_lds_b128)
#define ASYNC_LDS 1
#endif
#if __has_builtin(__builtin_amdgcn_s_wait_asynccnt)
#define HAVE_WAIT_ASYNC 1
#endif
#endif

__device__ __forceinline__ void async_cp16(const __bf16* g, __bf16* l) {
#if defined(ASYNC_LDS)
    __builtin_amdgcn_global_load_async_to_lds_b128(
        (__attribute__((address_space(1))) v4i*)g,
        (__attribute__((address_space(3))) v4i*)l, 0, 0);
#else
    *(uint4*)l = *(const uint4*)g;
#endif
}

template <int N>
__device__ __forceinline__ void wait_async() {
#if defined(HAVE_WAIT_ASYNC)
    __builtin_amdgcn_s_wait_asynccnt((unsigned short)N);
#elif defined(ASYNC_LDS)
    asm volatile("s_wait_asynccnt %0" :: "n"(N) : "memory");
#endif
}

__device__ __forceinline__ float fast_sigmoid(float x) {
    return 1.0f / (1.0f + __expf(-x));
}
__device__ __forceinline__ float fast_tanh(float x) {
    float xc = fminf(fmaxf(x, -15.0f), 15.0f);
    float e  = __expf(2.0f * xc);
    return (e - 1.0f) / (e + 1.0f);
}

// ---------------------------------------------------------------------------
// Kernel 1: combined = concat(input, h_prev) -> bf16, row-major (BSZ x DIN)
// ---------------------------------------------------------------------------
__global__ __launch_bounds__(256) void cvt_combined(const float* __restrict__ xin,
                                                    const float* __restrict__ hp,
                                                    __bf16* __restrict__ out) {
    size_t t    = (size_t)blockIdx.x * 256 + threadIdx.x;
    size_t base = t * 8;                       // 8 elems / thread
    size_t row  = base >> 11;                  // / DIN
    size_t col  = base & (DIN - 1);
    const float* src = (col < INSZ) ? (xin + row * INSZ + col)
                                    : (hp  + row * HID  + (col - INSZ));
    float4 a = *(const float4*)src;
    float4 b = *(const float4*)(src + 4);
    union { __bf16 v[8]; uint4 u; } o;
    o.v[0] = (__bf16)a.x; o.v[1] = (__bf16)a.y; o.v[2] = (__bf16)a.z; o.v[3] = (__bf16)a.w;
    o.v[4] = (__bf16)b.x; o.v[5] = (__bf16)b.y; o.v[6] = (__bf16)b.z; o.v[7] = (__bf16)b.w;
    *(uint4*)(out + base) = o.u;
}

// ---------------------------------------------------------------------------
// Kernel 2: Wb = [W_lstm ; W_level] -> bf16, row-major (WROWS x DIN)
// ---------------------------------------------------------------------------
__global__ __launch_bounds__(256) void cvt_weights(const float* __restrict__ Wl,
                                                   const float* __restrict__ Wv,
                                                   __bf16* __restrict__ out) {
    size_t t    = (size_t)blockIdx.x * 256 + threadIdx.x;
    size_t base = t * 8;
    const size_t LSTM_ELEMS = (size_t)(4 * HID) * DIN;
    const float* src = (base < LSTM_ELEMS) ? (Wl + base) : (Wv + (base - LSTM_ELEMS));
    float4 a = *(const float4*)src;
    float4 b = *(const float4*)(src + 4);
    union { __bf16 v[8]; uint4 u; } o;
    o.v[0] = (__bf16)a.x; o.v[1] = (__bf16)a.y; o.v[2] = (__bf16)a.z; o.v[3] = (__bf16)a.w;
    o.v[4] = (__bf16)b.x; o.v[5] = (__bf16)b.y; o.v[6] = (__bf16)b.z; o.v[7] = (__bf16)b.w;
    *(uint4*)(out + base) = o.u;
}

// ---------------------------------------------------------------------------
// Kernel 3: level GEMM (8192x2048 @ 2048x128) + softmax + cumsum -> IH, FH
// One block: 16 rows x all 128 level columns; 8 waves, one 16x16 tile each.
// ---------------------------------------------------------------------------
__global__ __launch_bounds__(256) void level_kernel(const __bf16* __restrict__ Cb,
                                                    const __bf16* __restrict__ Wb,
                                                    float* __restrict__ IH,
                                                    float* __restrict__ FH) {
    __shared__ __align__(16) __bf16 As[16][LDSPAD];
    __shared__ __align__(16) __bf16 Bs[128][LDSPAD];
    __shared__ float Lb[16][128];

    const int tid  = threadIdx.x;
    const int wave = tid >> 5;
    const int lane = tid & 31;
    const int rowBase = blockIdx.x * 16;
    const __bf16* Wlvl = Wb + (size_t)(4 * HID) * DIN;

    v8f acc = {};
    for (int k0 = 0; k0 < DIN; k0 += 32) {
        if (tid < 64) {                                  // A tile: 64 x 16B
            int r = tid >> 2, ch = tid & 3;
            *(uint4*)&As[r][ch * 8] =
                *(const uint4*)(Cb + (size_t)(rowBase + r) * DIN + k0 + ch * 8);
        }
        {                                                // B tile: 256 x 32B
            int r = tid >> 1, ch = tid & 1;
            const __bf16* src = Wlvl + (size_t)r * DIN + k0 + ch * 16;
            *(uint4*)&Bs[r][ch * 16]     = *(const uint4*)src;
            *(uint4*)&Bs[r][ch * 16 + 8] = *(const uint4*)(src + 8);
        }
        __syncthreads();

        const int m  = lane & 15;
        const int kb = (lane >> 4) * 8;
        const int ko = (lane >> 4) * 16;
        union { uint4 u[2]; v16bf v; } af, bfm;
        af.u[0] = *(const uint4*)&As[m][kb];
        af.u[1] = *(const uint4*)&As[m][kb + 16];
        int brow = wave * 16 + (lane & 15);
        bfm.u[0] = *(const uint4*)&Bs[brow][kb];
        bfm.u[1] = *(const uint4*)&Bs[brow][kb + 16];
        acc = __builtin_amdgcn_wmma_f32_16x16x32_bf16(false, af.v, false, bfm.v,
                                                      (short)0, acc, false, false);
        __syncthreads();
    }

    {   // scatter C tile: vgpr r -> M = r + 8*(lane>=16), N = lane&15
        const int cn   = lane & 15;
        const int moff = (lane >> 4) * 8;
        for (int r = 0; r < 8; ++r)
            Lb[moff + r][wave * 16 + cn] = acc[r];
    }
    __syncthreads();

    if (tid < 32) {   // 16 rows x {i-half(reverse), f-half(forward)}
        int r    = tid >> 1;
        int half = tid & 1;
        const float* src = &Lb[r][half * 64];
        float mx = -3.4e38f;
        for (int l = 0; l < 64; ++l) mx = fmaxf(mx, src[l]);
        float s = 0.0f;
        for (int l = 0; l < 64; ++l) s += __expf(src[l] - mx);
        float inv = 1.0f / s;
        size_t obase = (size_t)(rowBase + r) * LVL;
        if (half == 1) {
            float run = 0.0f;
            for (int l = 0; l < 64; ++l) { run += __expf(src[l] - mx) * inv; FH[obase + l] = run; }
        } else {
            float run = 0.0f;
            for (int l = 63; l >= 0; --l) { run += __expf(src[l] - mx) * inv; IH[obase + l] = run; }
        }
    }
}

// ---------------------------------------------------------------------------
// Kernel 4: main LSTM GEMM + fused ON-LSTM epilogue.
// Block = 256 threads = 8 waves (4 along M x 2 along N).
// Block tile: 128 rows x 32 hidden cols, computed for ALL FOUR gates
// (cols j, j+1024, j+2048, j+3072) sharing A fragments.
// Double-buffered LDS fed by async global->LDS (ASYNCcnt) copies.
// Per wave per K-step: 2 A frags, 4 B frags, 8 WMMAs.
// ---------------------------------------------------------------------------
__global__ __launch_bounds__(256) void lstm_main(const __bf16* __restrict__ Cb,
                                                 const __bf16* __restrict__ Wb,
                                                 const float* __restrict__ c_prev,
                                                 const float* __restrict__ IH,
                                                 const float* __restrict__ FH,
                                                 float* __restrict__ out) {
    __shared__ __align__(16) __bf16 As[2][BM][LDSPAD];    // 2 x 128 x 32K
    __shared__ __align__(16) __bf16 Bs[2][128][LDSPAD];   // 2 x (4 gates * 32 n) x 32K

    const int tid   = threadIdx.x;
    const int wave  = tid >> 5;
    const int lane  = tid & 31;
    const int waveM = wave >> 1;         // 0..3 -> 32 rows each
    const int waveN = wave & 1;          // 0..1 -> 16 cols each
    const int rowBase = blockIdx.y * BM;
    const int colBase = blockIdx.x * BN;

    v8f acc[2][4] = {};                  // [m-subtile][gate]

    // per-thread fill roles: row = tid>>1 (0..127), 16-element chunk = (tid&1)*16
    const int fr = tid >> 1;
    const int fc = (tid & 1) * 16;
    const int bgate = fr >> 5, bn = fr & 31;
    const size_t aOff = (size_t)(rowBase + fr) * DIN + fc;
    const size_t bOff = (size_t)(bgate * HID + colBase + bn) * DIN + fc;

    // prologue: issue tile 0 into buffer 0 (4 async ops / thread)
    async_cp16(Cb + aOff,     &As[0][fr][fc]);
    async_cp16(Cb + aOff + 8, &As[0][fr][fc + 8]);
    async_cp16(Wb + bOff,     &Bs[0][fr][fc]);
    async_cp16(Wb + bOff + 8, &Bs[0][fr][fc + 8]);

    const int m  = lane & 15;
    const int kb = (lane >> 4) * 8;
    const int ko = (lane >> 4) * 16;

    for (int kt = 0; kt < NKT; ++kt) {
        const int cur = kt & 1;
        if (kt + 1 < NKT) {
            const int nxt = cur ^ 1;
            const size_t k1 = (size_t)(kt + 1) * KT;
            async_cp16(Cb + aOff + k1,     &As[nxt][fr][fc]);
            async_cp16(Cb + aOff + k1 + 8, &As[nxt][fr][fc + 8]);
            async_cp16(Wb + bOff + k1,     &Bs[nxt][fr][fc]);
            async_cp16(Wb + bOff + k1 + 8, &Bs[nxt][fr][fc + 8]);
            wait_async<4>();             // in-order: current tile's 4 ops done
        } else {
            wait_async<0>();
        }
        __syncthreads();                 // all waves' tile data visible in LDS

        union { uint4 u[2]; v16bf v; } af0, af1, bfm;
        af0.u[0] = *(const uint4*)&As[cur][waveM * 32 + m][kb];
        af0.u[1] = *(const uint4*)&As[cur][waveM * 32 + m][kb + 16];
        af1.u[0] = *(const uint4*)&As[cur][waveM * 32 + 16 + m][kb];
        af1.u[1] = *(const uint4*)&As[cur][waveM * 32 + 16 + m][kb + 16];
        #pragma unroll
        for (int g = 0; g < 4; ++g) {
            int brow = g * 32 + waveN * 16 + (lane & 15);
            bfm.u[0] = *(const uint4*)&Bs[cur][brow][kb];
            bfm.u[1] = *(const uint4*)&Bs[cur][brow][kb + 16];
            acc[0][g] = __builtin_amdgcn_wmma_f32_16x16x32_bf16(false, af0.v, false, bfm.v,
                                                                (short)0, acc[0][g], false, false);
            acc[1][g] = __builtin_amdgcn_wmma_f32_16x16x32_bf16(false, af1.v, false, bfm.v,
                                                                (short)0, acc[1][g], false, false);
        }
        __syncthreads();                 // protect buffer 'cur' before refill at kt+1
    }

    // ---- fused epilogue ----
    const int cn   = lane & 15;
    const int moff = (lane >> 4) * 8;
    const int j    = colBase + waveN * 16 + cn;     // hidden index
    const int l    = j >> 4;                        // level index (repeat=16)
    #pragma unroll
    for (int mi = 0; mi < 2; ++mi) {
        #pragma unroll
        for (int r = 0; r < 8; ++r) {
            const int row = rowBase + waveM * 32 + mi * 16 + moff + r;
            float iv = fast_sigmoid(acc[mi][0][r]);
            float fv = fast_sigmoid(acc[mi][1][r]);
            float ov = fast_sigmoid(acc[mi][2][r]);
            float gv = fast_tanh(acc[mi][3][r]);
            float ih = IH[(size_t)row * LVL + l];
            float fh = FH[(size_t)row * LVL + l];
            float cp = c_prev[(size_t)row * HID + j];
            float w  = ih * fh;
            float c  = w * (fv * cp + iv * gv) + (fh - w) * cp + (ih - w) * gv;
            float h  = ov * fast_tanh(c);
            out[(size_t)row * HID + j]                     = h;
            out[(size_t)BSZ * HID + (size_t)row * HID + j] = c;
        }
    }
}

// ---------------------------------------------------------------------------
extern "C" void kernel_launch(void* const* d_in, const int* in_sizes, int n_in,
                              void* d_out, int out_size, void* d_ws, size_t ws_size,
                              hipStream_t stream) {
    const float* input  = (const float*)d_in[0];   // (8192,1024)
    const float* h_prev = (const float*)d_in[1];   // (8192,1024)
    const float* c_prev = (const float*)d_in[2];   // (8192,1024)
    const float* W_lstm = (const float*)d_in[3];   // (4096,2048)
    const float* W_lvl  = (const float*)d_in[4];   // (128,2048)
    float* out = (float*)d_out;                    // h then c, each 8192*1024

    __bf16* Cb = (__bf16*)d_ws;                                 // 8192*2048 bf16
    __bf16* Wb = Cb + (size_t)BSZ * DIN;                        // 4224*2048 bf16
    float*  IH = (float*)(Wb + (size_t)WROWS * DIN);            // 8192*64 f32
    float*  FH = IH + (size_t)BSZ * LVL;                        // 8192*64 f32

    cvt_combined<<<(BSZ * DIN) / (256 * 8), 256, 0, stream>>>(input, h_prev, Cb);
    cvt_weights <<<(WROWS * DIN) / (256 * 8), 256, 0, stream>>>(W_lstm, W_lvl, Wb);
    level_kernel<<<BSZ / 16, 256, 0, stream>>>(Cb, Wb, IH, FH);
    dim3 grid(HID / BN, BSZ / BM);
    lstm_main<<<grid, 256, 0, stream>>>(Cb, Wb, c_prev, IH, FH, out);
}
